// MultiHeadAttention_48627619725735
// MI455X (gfx1250) — matrix operand order repaired and checked by reading, not compiled, not run on hardware
//
#include <hip/hip_runtime.h>
#include <math.h>

typedef _Float16 v16h __attribute__((ext_vector_type(16)));
typedef _Float16 v8h  __attribute__((ext_vector_type(8)));
typedef float    v8f  __attribute__((ext_vector_type(8)));
typedef float    v2f  __attribute__((ext_vector_type(2)));

#define IMG_H 128
#define IMG_W 48
#define CIN   64
#define DMODEL 32
#define DH    4
// per (head, block): 3072 queries, 4096 valid keys (128 rows x 32 cols)
#define NQ    3072
#define NK    4096
#define QTILES 192        // 3072 / 16
#define KITERS 128        // 4096 / 32

// Q scale: dh^-0.5 * log2(e)  -> logits land directly in the base-2 domain,
// so softmax uses raw v_exp_f32 (which is exp2) with no per-element rescale.
#define QSCALE 0.72134752044448170368f   // 0.5 * 1.4426950408889634

// ---------------------------------------------------------------------------
// Kernel 1: fused QKV 3x3 conv projections (f32). Q/K stay f32 (feed the
// 16x16x4 f32 WMMA exactly at d=4); V is emitted f16 transposed for the
// 16x16x32 f16 PV WMMA.
// ---------------------------------------------------------------------------
__global__ __launch_bounds__(256) void qkv_conv_kernel(
    const float* __restrict__ x,
    const float* __restrict__ wq, const float* __restrict__ bq,
    const float* __restrict__ wk, const float* __restrict__ bk,
    const float* __restrict__ wv, const float* __restrict__ bv,
    float* __restrict__ Q, float* __restrict__ K,
    _Float16* __restrict__ Vt) {
  int idx = blockIdx.x * 256 + threadIdx.x;
  if (idx >= IMG_H * IMG_W * DMODEL) return;
  int j = idx & 31;                 // output channel 0..31
  int c = (idx >> 5) % IMG_W;
  int r = idx / (32 * IMG_W);

  float aq = bq[j], ak = bk[j], av = bv[j];
  #pragma unroll
  for (int ky = 0; ky < 3; ++ky) {
    int rr = r + ky - 1;
    if (rr < 0 || rr >= IMG_H) continue;
    #pragma unroll
    for (int kx = 0; kx < 3; ++kx) {
      int cc = c + kx - 1;
      if (cc < 0 || cc >= IMG_W) continue;
      const float* xp  = x  + ((size_t)rr * IMG_W + cc) * CIN;
      const float* wqp = wq + ((ky * 3 + kx) * CIN) * DMODEL + j;
      const float* wkp = wk + ((ky * 3 + kx) * CIN) * DMODEL + j;
      const float* wvp = wv + ((ky * 3 + kx) * CIN) * DMODEL + j;
      for (int ci = 0; ci < CIN; ++ci) {
        float xv = xp[ci];
        aq = fmaf(xv, wqp[ci * DMODEL], aq);
        ak = fmaf(xv, wkp[ci * DMODEL], ak);
        av = fmaf(xv, wvp[ci * DMODEL], av);
      }
    }
  }

  int h = j >> 2, d = j & 3;
  // Q: block g = c/24, local query index = r*24 + c%24
  {
    int g  = c / 24;
    int ql = r * 24 + (c % 24);
    Q[(((size_t)(h * 2 + g)) * NQ + ql) * DH + d] = aq * QSCALE;
  }
  // K/V valid-key subgrids: block 0 covers image cols [0,32), block 1 [16,48)
  if (c < 32) {
    int key = r * 32 + c;
    size_t hg = (size_t)(h * 2 + 0);
    K [(hg * NK + key) * DH + d] = ak;
    Vt[(hg * DH + d) * NK + key] = (_Float16)av;
  }
  if (c >= 16) {
    int key = r * 32 + (c - 16);
    size_t hg = (size_t)(h * 2 + 1);
    K [(hg * NK + key) * DH + d] = ak;
    Vt[(hg * DH + d) * NK + key] = (_Float16)av;
  }
}

// ---------------------------------------------------------------------------
// Kernel 2: two-pass flash attention, one wave per 16-query tile.
// S^T = K x Q^T via v_wmma_f32_16x16x4_f32 (contraction = d = 4 exactly:
// all 32 lanes carry payload, 2-VGPR fragments, no padding, no predication).
// The exp'd S^T accumulator is bit-exact the A-fragment of the f16
// O = P x V WMMA (contraction = 32 keys). Softmax stats are per-lane and
// entirely in the base-2 domain (raw v_exp_f32).
// ---------------------------------------------------------------------------
__global__ __launch_bounds__(256) void local_attn_kernel(
    const float* __restrict__ Q, const float* __restrict__ K,
    const _Float16* __restrict__ Vt, float* __restrict__ O) {
  int lane = threadIdx.x & 31;
  int w    = blockIdx.x * 8 + (threadIdx.x >> 5);   // 0..3071
  int hg   = w / QTILES;                            // head*2 + block
  int qt   = w % QTILES;

  const float*    Qp = Q  + (size_t)hg * NQ * DH;
  const float*    Kp = K  + (size_t)hg * NK * DH;
  const _Float16* Vp = Vt + (size_t)hg * DH * NK;
  float*          Op = O  + (size_t)hg * NQ * DH;

  int col  = lane & 15;   // query column owned by this lane / key row in tile
  int half = lane >> 4;   // selects contraction pair (d 0,1 vs d 2,3)

  // Q B-fragment (4x16 f32): lane = query column, 2 VGPRs = d(2h), d(2h+1)
  v2f qb = *(const v2f*)(Qp + (size_t)(qt * 16 + col) * DH + 2 * half);

  // K A-fragment base: lane = key row in tile, 2 VGPRs = d(2h), d(2h+1)
  const float* kbase = Kp + (size_t)col * DH + 2 * half;

  // ---------------- Pass A: exact per-query max (base-2 logits) ----------------
  float mx = -INFINITY;
  for (int kt = 0; kt < KITERS; ++kt) {
    int k0 = kt * 32;
    v2f ka0 = *(const v2f*)(kbase + (size_t)k0 * DH);
    v2f ka1 = *(const v2f*)(kbase + (size_t)(k0 + 16) * DH);
    v8f z = {};
    v8f s0 = __builtin_amdgcn_wmma_f32_16x16x4_f32(false, ka0, false, qb,
                                                   (short)0, z, false, false);
    v8f s1 = __builtin_amdgcn_wmma_f32_16x16x4_f32(false, ka1, false, qb,
                                                   (short)0, z, false, false);
    #pragma unroll
    for (int r = 0; r < 8; ++r)
      mx = fmaxf(mx, fmaxf(s0[r], s1[r]));
  }
  mx = fmaxf(mx, __shfl_xor(mx, 16, 32));   // combine the two key-row halves

  // ---------------- Pass B: p = exp2(s - m), l, O = P x V ----------------
  // V B-fragment: output column n only reads B column n, so lanes with
  // col >= 4 load a wrapped copy (col & 3) -- their output columns are
  // never stored. No zeroing, no predication.
  const _Float16* vbase = Vp + (size_t)(col & 3) * NK + 8 * half;

  float lsum = 0.0f;
  v8f oacc = {};
  for (int kt = 0; kt < KITERS; ++kt) {
    int k0 = kt * 32;
    v2f ka0 = *(const v2f*)(kbase + (size_t)k0 * DH);
    v2f ka1 = *(const v2f*)(kbase + (size_t)(k0 + 16) * DH);
    const v8h* vp = (const v8h*)(vbase + k0);
    v8h lo = vp[0], hi = vp[2];

    v8f z = {};
    v8f s0 = __builtin_amdgcn_wmma_f32_16x16x4_f32(false, ka0, false, qb,
                                                   (short)0, z, false, false);
    v8f s1 = __builtin_amdgcn_wmma_f32_16x16x4_f32(false, ka1, false, qb,
                                                   (short)0, z, false, false);

    // exp2 + per-lane partial sum; accumulator layout == PV A-fragment layout
    v16h ap;
    #pragma unroll
    for (int r = 0; r < 8; ++r) {
      float e0 = __builtin_amdgcn_exp2f(s0[r] - mx);
      float e1 = __builtin_amdgcn_exp2f(s1[r] - mx);
      lsum += e0 + e1;
      ap[r]     = (_Float16)e0;   // keys 8*half + r        (tile 0)
      ap[8 + r] = (_Float16)e1;   // keys 16 + 8*half + r   (tile 1)
    }

    v16h bvf;
    #pragma unroll
    for (int i = 0; i < 8; ++i) { bvf[i] = lo[i]; bvf[8 + i] = hi[i]; }

    oacc = __builtin_amdgcn_wmma_f32_16x16x32_f16(false, ap, false, bvf,
                                                  (short)0, oacc, false, false);
  }

  lsum += __shfl_xor(lsum, 16, 32);   // full softmax denominator per query
  float rinv = 1.0f / lsum;           // one divide per lane

  // O rows are queries 8*half + r; fetch their reciprocal denominators
  // (all lanes participate in the shuffles; the store is predicated after).
  float rq[8];
  #pragma unroll
  for (int r = 0; r < 8; ++r)
    rq[r] = __shfl(rinv, half * 8 + r, 32);

  if (col < 4) {
    #pragma unroll
    for (int r = 0; r < 8; ++r) {
      int q = qt * 16 + half * 8 + r;
      Op[(size_t)q * DH + col] = oacc[r] * rq[r];
    }
  }
}

// ---------------------------------------------------------------------------
// Kernel 3: 3x3 output conv (32 -> 64), reading attention output via index
// math (head-block layout -> NHWC), no bias.
// ---------------------------------------------------------------------------
__global__ __launch_bounds__(256) void out_conv_kernel(
    const float* __restrict__ Ob, const float* __restrict__ wo,
    float* __restrict__ out) {
  int idx = blockIdx.x * 256 + threadIdx.x;
  if (idx >= IMG_H * IMG_W * 64) return;
  int co = idx & 63;
  int c  = (idx >> 6) % IMG_W;
  int r  = idx / (64 * IMG_W);

  float acc = 0.0f;
  #pragma unroll
  for (int ky = 0; ky < 3; ++ky) {
    int rr = r + ky - 1;
    if (rr < 0 || rr >= IMG_H) continue;
    #pragma unroll
    for (int kx = 0; kx < 3; ++kx) {
      int cc = c + kx - 1;
      if (cc < 0 || cc >= IMG_W) continue;
      int g  = cc / 24;
      int ql = rr * 24 + (cc % 24);
      const float* wp = wo + ((ky * 3 + kx) * DMODEL) * 64 + co;
      for (int ci = 0; ci < DMODEL; ++ci) {
        int h = ci >> 2, d = ci & 3;
        float ov = Ob[(((size_t)(h * 2 + g)) * NQ + ql) * DH + d];
        acc = fmaf(ov, wp[ci * 64], acc);
      }
    }
  }
  out[idx] = acc;
}

// ---------------------------------------------------------------------------
// Launch: inputs x, wq, bq, wk, bk, wv, bv, wo (all f32). Output f32
// (128,48,64). Workspace: Q f32 | K f32 | V^T f16 | O f32  (~3.1 MB).
// ---------------------------------------------------------------------------
extern "C" void kernel_launch(void* const* d_in, const int* in_sizes, int n_in,
                              void* d_out, int out_size, void* d_ws, size_t ws_size,
                              hipStream_t stream) {
  const float* x  = (const float*)d_in[0];
  const float* wq = (const float*)d_in[1];
  const float* bq = (const float*)d_in[2];
  const float* wk = (const float*)d_in[3];
  const float* bk = (const float*)d_in[4];
  const float* wv = (const float*)d_in[5];
  const float* bv = (const float*)d_in[6];
  const float* wo = (const float*)d_in[7];

  char* ws = (char*)d_ws;
  float*    Q  = (float*)   (ws);                       // 16*3072*4*4  =  786432 B
  float*    K  = (float*)   (ws + 786432);              // 16*4096*4*4  = 1048576 B
  _Float16* Vt = (_Float16*)(ws + 1835008);             // 16*4*4096*2  =  524288 B
  float*    Ob = (float*)   (ws + 2359296);             // 16*3072*4*4  =  786432 B

  qkv_conv_kernel<<<(IMG_H * IMG_W * DMODEL + 255) / 256, 256, 0, stream>>>(
      x, wq, bq, wk, bk, wv, bv, Q, K, Vt);

  // 3072 waves, 8 waves (256 threads) per block
  local_attn_kernel<<<384, 256, 0, stream>>>(Q, K, Vt, Ob);

  out_conv_kernel<<<(IMG_H * IMG_W * 64 + 255) / 256, 256, 0, stream>>>(
      Ob, wo, (float*)d_out);
}
